// NNUE_37838661878266
// MI455X (gfx1250) — matrix operand order repaired and checked by reading, not compiled, not run on hardware
//
#include <hip/hip_runtime.h>

#define F_DIM 41024
#define M_DIM 512
#define ROWS  8192       // B * P
#define NB    4096       // batches

typedef __bf16 bf16_t;
typedef __attribute__((ext_vector_type(16))) __bf16 v16bf;
typedef __attribute__((ext_vector_type(8)))  __bf16 v8bf;
typedef __attribute__((ext_vector_type(8)))  float  v8f;
typedef __attribute__((ext_vector_type(4)))  float  v4f;

__device__ __forceinline__ float clamp01(float x) {
    return __builtin_fminf(__builtin_fmaxf(x, 0.0f), 1.0f);
}
__device__ __forceinline__ void sfence() { __builtin_amdgcn_sched_barrier(0); }

// ---------------------------------------------------------------------------
// Kernel 1: convert W_ft (f32, [512, F]) -> bf16 in workspace (~42 MB).
// Source read once -> non-temporal; bf16 result is the hot L2-resident GEMM
// operand, so stores keep the default temporal policy.
// ---------------------------------------------------------------------------
__global__ void __launch_bounds__(256)
wft_to_bf16(const float* __restrict__ src, bf16_t* __restrict__ dst, int n8) {
    int idx    = blockIdx.x * blockDim.x + threadIdx.x;
    int stride = gridDim.x * blockDim.x;
    for (int i = idx; i < n8; i += stride) {
        const v4f* s = (const v4f*)(src + (size_t)i * 8);
        v4f a = __builtin_nontemporal_load(s);
        v4f b = __builtin_nontemporal_load(s + 1);
        v8bf o;
        o[0] = (bf16_t)a[0]; o[1] = (bf16_t)a[1];
        o[2] = (bf16_t)a[2]; o[3] = (bf16_t)a[3];
        o[4] = (bf16_t)b[0]; o[5] = (bf16_t)b[1];
        o[6] = (bf16_t)b[2]; o[7] = (bf16_t)b[3];
        *(v8bf*)(dst + (size_t)i * 8) = o;
    }
}

// Raw A loads (f32, non-temporal: features are a 1.34 GB single-use stream and
// must not evict W_ft from the 192 MB L2). k order used here for A (B matches), per lane:
// elems 0..15 = K k0+16h+(0..15).
__device__ __forceinline__ void load_a_raw(v4f (&af)[4],
                                           const float* __restrict__ aptr, int k0) {
    af[0] = __builtin_nontemporal_load((const v4f*)(aptr + k0));
    af[1] = __builtin_nontemporal_load((const v4f*)(aptr + k0 + 4));
    af[2] = __builtin_nontemporal_load((const v4f*)(aptr + k0 + 8));
    af[3] = __builtin_nontemporal_load((const v4f*)(aptr + k0 + 12));
}

__device__ __forceinline__ v16bf cvt_a(const v4f (&af)[4]) {
    v16bf a;
    a[0]  = (bf16_t)af[0][0]; a[1]  = (bf16_t)af[0][1];
    a[2]  = (bf16_t)af[0][2]; a[3]  = (bf16_t)af[0][3];
    a[4]  = (bf16_t)af[1][0]; a[5]  = (bf16_t)af[1][1];
    a[6]  = (bf16_t)af[1][2]; a[7]  = (bf16_t)af[1][3];
    a[8]  = (bf16_t)af[2][0]; a[9]  = (bf16_t)af[2][1];
    a[10] = (bf16_t)af[2][2]; a[11] = (bf16_t)af[2][3];
    a[12] = (bf16_t)af[3][0]; a[13] = (bf16_t)af[3][1];
    a[14] = (bf16_t)af[3][2]; a[15] = (bf16_t)af[3][3];
    return a;
}

__device__ __forceinline__ v16bf load_b_frag_f32(const float* __restrict__ bp) {
    v4f g0 = *(const v4f*)bp;
    v4f g1 = *(const v4f*)(bp + 4);
    v4f g2 = *(const v4f*)(bp + 8);
    v4f g3 = *(const v4f*)(bp + 12);
    v16bf b;
    b[0]  = (bf16_t)g0[0]; b[1]  = (bf16_t)g0[1];
    b[2]  = (bf16_t)g0[2]; b[3]  = (bf16_t)g0[3];
    b[4]  = (bf16_t)g1[0]; b[5]  = (bf16_t)g1[1];
    b[6]  = (bf16_t)g1[2]; b[7]  = (bf16_t)g1[3];
    b[8]  = (bf16_t)g2[0]; b[9]  = (bf16_t)g2[1];
    b[10] = (bf16_t)g2[2]; b[11] = (bf16_t)g2[3];
    b[12] = (bf16_t)g3[0]; b[13] = (bf16_t)g3[1];
    b[14] = (bf16_t)g3[2]; b[15] = (bf16_t)g3[3];
    return b;
}

// B tile: uniform per-t base (wave-uniform -> SADDR) + one lane-varying
// 32-bit element offset; k advances on the uniform side.
template <bool BF>
__device__ __forceinline__ void load_b_tile(v16bf (&dst)[8],
                                            const bf16_t* const (&bbb)[8],
                                            const float*  const (&bbf)[8],
                                            int voff, int k0) {
#pragma unroll
    for (int t = 0; t < 8; ++t) {
        if constexpr (BF) dst[t] = *(const v16bf*)(bbb[t] + k0 + voff);
        else              dst[t] = load_b_frag_f32(bbf[t] + k0 + voff);
    }
}

__device__ __forceinline__ void wmma8(v8f (&acc)[8], const v16bf& a,
                                      const v16bf (&b)[8]) {
#pragma unroll
    for (int t = 0; t < 8; ++t)
        acc[t] = __builtin_amdgcn_wmma_f32_16x16x32_bf16(
            false, a, false, b[t], (short)0, acc[t], false, false);
}

// ---------------------------------------------------------------------------
// Kernel 2: fully fused NNUE forward.
//   Block tile: 32 rows (16 batches x 2 persp) x 512 cols (all of M).
//   8 waves: rowGroup = wave>>2 (0..1), colGroup = wave&3 (0..3).
//   Wave tile: 16 rows x 128 cols = 8 WMMA C tiles.
//   K pipeline (32-K phases, sched_barrier-fenced so the scheduler cannot
//   collapse it):  [wmma8 on B loaded 2 phases ago] -> [cvt A loaded 1 phase
//   ago] -> [issue A raw loads K+64] -> [issue B loads K+64 into the buffer
//   freed this phase].  A-waits imply loadcnt<=16 (B stays in flight); B-waits
//   are staggered with ~2 phases of slack.
// ---------------------------------------------------------------------------
template <bool B_IS_BF16>
__global__ void __launch_bounds__(256)
nnue_fused(const float*  __restrict__ feat,     // [8192, F] f32
           const float*  __restrict__ stm,      // [8192]
           const float*  __restrict__ wft_f32,  // [512, F] f32 (fallback)
           const bf16_t* __restrict__ wft_bf,   // [512, F] bf16 (fast path)
           const float*  __restrict__ b_ft,     // [512]
           const float*  __restrict__ W1,       // [32, 1024]
           const float*  __restrict__ b1,       // [32]
           const float*  __restrict__ W2,       // [32]
           const float*  __restrict__ b2,       // [1]
           float*        __restrict__ out)      // [4096]
{
    __shared__ float l1buf[16 * 1024];          // 64 KB of the 320 KB WGP LDS

    const int tid      = threadIdx.x;
    const int wave     = tid >> 5;
    const int lane     = tid & 31;
    const int l        = lane & 15;
    const int h        = lane >> 4;
    const int rowGroup = wave >> 2;             // 0..1
    const int colGroup = wave & 3;              // 0..3

    const int blockRow0 = blockIdx.x * 32;
    const int aRow      = blockRow0 + rowGroup * 16 + l;
    const int colBase   = colGroup * 128;

    v8f acc[8];
#pragma unroll
    for (int t = 0; t < 8; ++t)
#pragma unroll
        for (int r = 0; r < 8; ++r) acc[t][r] = 0.0f;

    const float* aptr = feat + (size_t)aRow * F_DIM + h * 16;

    // Wave-uniform per-tile B bases (-> SGPRs) + lane-varying element offset.
    const bf16_t* bbb[8];
    const float*  bbf[8];
#pragma unroll
    for (int t = 0; t < 8; ++t) {
        const size_t ubase = (size_t)(colBase + t * 16) * F_DIM;
        if constexpr (B_IS_BF16) { bbb[t] = wft_bf + ubase;  bbf[t] = nullptr; }
        else                     { bbf[t] = wft_f32 + ubase; bbb[t] = nullptr; }
    }
    const int voffB = l * F_DIM + h * 16;       // lane-varying, 32-bit

    v16bf b0[8], b1v[8], a;
    v4f   af[4];

    // Prologue: B(0) + A(0) direct; issue A(32) raw then B(32).
    load_b_tile<B_IS_BF16>(b0, bbb, bbf, voffB, 0);
    load_a_raw(af, aptr, 0);
    a = cvt_a(af);
    load_a_raw(af, aptr, 32);
    load_b_tile<B_IS_BF16>(b1v, bbb, bbf, voffB, 32);

#pragma unroll 1
    for (int p = 0; p < 640; ++p) {
        const int k0 = p * 64;
        // ---- even phase: compute K=k0 with b0 ----
        wmma8(acc, a, b0);
        sfence();
        a = cvt_a(af);                          // A(k0+32), loaded last phase
        sfence();
        load_a_raw(af, aptr, k0 + 64);          // A(k0+64) first (loadcnt<=16)
        sfence();
        load_b_tile<B_IS_BF16>(b0, bbb, bbf, voffB, k0 + 64);
        sfence();
        // ---- odd phase: compute K=k0+32 with b1 ----
        wmma8(acc, a, b1v);
        sfence();
        a = cvt_a(af);                          // A(k0+64)
        sfence();
        load_a_raw(af, aptr, k0 + 96);          // A(k0+96)
        sfence();
        load_b_tile<B_IS_BF16>(b1v, bbb, bbf, voffB, k0 + 96);
        sfence();
    }
    // Epilogue phases: K=40960 (b0), K=40992 (b1); no further prefetch.
    wmma8(acc, a, b0);
    a = cvt_a(af);
    wmma8(acc, a, b1v);

    // ---- Epilogue: bias + stm scale + clamp -> LDS l1_x -----------------
    float stmv[8];
#pragma unroll
    for (int r = 0; r < 8; ++r)
        stmv[r] = stm[blockRow0 + rowGroup * 16 + h * 8 + r];

#pragma unroll
    for (int t = 0; t < 8; ++t) {
        const int   m    = colBase + t * 16 + l;
        const float bias = b_ft[m];
#pragma unroll
        for (int r = 0; r < 8; ++r) {
            const int rowInBlock = rowGroup * 16 + h * 8 + r;   // 0..31
            const float v = clamp01(stmv[r] * (acc[t][r] + bias));
            l1buf[(rowInBlock >> 1) * 1024 + (rowInBlock & 1) * 512 + m] = v;
        }
    }
    __syncthreads();

    // ---- Layers 1 & 2: wave handles batches 2*wave, 2*wave+1; lane = neuron
    const int    j     = lane;
    const float* w1row = W1 + j * 1024;
    const float* x0    = l1buf + (wave * 2) * 1024;
    const float* x1    = x0 + 1024;
    float s0 = 0.0f, s1 = 0.0f;
#pragma unroll 4
    for (int i = 0; i < 1024; i += 4) {
        v4f wv = *(const v4f*)(w1row + i);
        v4f a0 = *(const v4f*)(x0 + i);
        v4f a1 = *(const v4f*)(x1 + i);
        s0 = __builtin_fmaf(a0[0], wv[0], s0);
        s0 = __builtin_fmaf(a0[1], wv[1], s0);
        s0 = __builtin_fmaf(a0[2], wv[2], s0);
        s0 = __builtin_fmaf(a0[3], wv[3], s0);
        s1 = __builtin_fmaf(a1[0], wv[0], s1);
        s1 = __builtin_fmaf(a1[1], wv[1], s1);
        s1 = __builtin_fmaf(a1[2], wv[2], s1);
        s1 = __builtin_fmaf(a1[3], wv[3], s1);
    }
    s0 = clamp01(s0 + b1[j]);
    s1 = clamp01(s1 + b1[j]);
    float p0 = s0 * W2[j];
    float p1 = s1 * W2[j];
#pragma unroll
    for (int off = 16; off > 0; off >>= 1) {
        p0 += __shfl_xor(p0, off, 32);
        p1 += __shfl_xor(p1, off, 32);
    }
    if (lane == 0) {
        const int gb = blockIdx.x * 16 + wave * 2;
        const float bb = b2[0];
        out[gb]     = p0 + bb;
        out[gb + 1] = p1 + bb;
    }
}

// ---------------------------------------------------------------------------
extern "C" void kernel_launch(void* const* d_in, const int* in_sizes, int n_in,
                              void* d_out, int out_size, void* d_ws, size_t ws_size,
                              hipStream_t stream) {
    (void)in_sizes; (void)n_in; (void)out_size;
    const float* feat = (const float*)d_in[0];   // [4096, 2, F]
    const float* stm  = (const float*)d_in[1];   // [4096, 2, 1]
    const float* wft  = (const float*)d_in[2];   // [512, F]
    const float* bft  = (const float*)d_in[3];   // [512]
    const float* W1   = (const float*)d_in[4];   // [32, 1024]
    const float* b1   = (const float*)d_in[5];   // [32]
    const float* W2   = (const float*)d_in[6];   // [1, 32]
    const float* b2   = (const float*)d_in[7];   // [1]
    float* out = (float*)d_out;

    const size_t wft_elems = (size_t)M_DIM * F_DIM;
    const size_t wft_bytes = wft_elems * sizeof(bf16_t);   // ~42 MB

    if (ws_size >= wft_bytes) {
        bf16_t* wftb = (bf16_t*)d_ws;
        const int n8 = (int)(wft_elems / 8);
        int cblocks = (n8 + 255) / 256;
        if (cblocks > 8192) cblocks = 8192;    // grid-stride covers the rest
        wft_to_bf16<<<cblocks, 256, 0, stream>>>(wft, wftb, n8);
        nnue_fused<true><<<ROWS / 32, 256, 0, stream>>>(
            feat, stm, nullptr, wftb, bft, W1, b1, W2, b2, out);
    } else {
        // Fallback: W_ft stays f32 (still L2-resident), converted in-register.
        nnue_fused<false><<<ROWS / 32, 256, 0, stream>>>(
            feat, stm, wft, nullptr, bft, W1, b1, W2, b2, out);
    }
}
